// LorentzConv2d_56788057588106
// MI455X (gfx1250) — matrix operand order repaired and checked by reading, not compiled, not run on hardware
//
#include <hip/hip_runtime.h>

// ---------------------------------------------------------------------------
// LorentzConv2d on gfx1250: split-bf16 WMMA GEMM
//   feat [131072 x 576(padded)] @ W' [576 x 64], fp32 accumulate.
//   x*w ~= x_hi*w_hi + x_hi*w_lo + x_lo*w_hi  (3x v_wmma_f32_16x16x32_bf16)
// ---------------------------------------------------------------------------

typedef __attribute__((ext_vector_type(16))) __bf16 v16bf;
typedef __attribute__((ext_vector_type(8)))  __bf16 v8bf;
typedef __attribute__((ext_vector_type(8)))  float  v8f;
typedef __attribute__((ext_vector_type(4)))  float  v4f;

#define HALO_STRIDE 72          // padded bf16 channel stride (144B) -> conflict-free ds_load_b128
#define NCHUNK      18          // K = 9 taps * 64 ch = 576 = 18 * 32
#define B_ELEMS     (NCHUNK * 4 * 32 * 16)   // bf16 elems per (hi or lo) weight panel

__device__ __forceinline__ void bf_split(float x, __bf16& hi, __bf16& lo) {
  __bf16 h = (__bf16)x;          // RNE to bf16
  hi = h;
  lo = (__bf16)(x - (float)h);   // residual
}

// ---------------------------------------------------------------------------
// Repack weight [64 x 568] into WMMA B-fragment order, K padded to 576:
//   k = tap*64 + c ; c==0 -> (tap==0 ? W[n][0] : 0) ; c>=1 -> W[n][1+tap*63+c-1]
// Layout: [chunk*4+nb][lane][16 bf16]  (lane<16: K=0..7,16..23, lane>=16: K=8..15,24..31)
// ---------------------------------------------------------------------------
__global__ void prep_B_kernel(const float* __restrict__ W,
                              unsigned short* __restrict__ Bhi_u,
                              unsigned short* __restrict__ Blo_u) {
  __bf16* Bhi = (__bf16*)Bhi_u;
  __bf16* Blo = (__bf16*)Blo_u;
  const int cb    = blockIdx.x;              // chunk*4 + nb, 72 blocks
  const int chunk = cb >> 2;
  const int nb    = cb & 3;
  const int lane  = threadIdx.x;             // 32 threads
  const int n     = nb * 16 + (lane & 15);   // output channel
  const int kb    = chunk * 32 + ((lane & 16) ? 8 : 0);
#pragma unroll
  for (int i = 0; i < 16; ++i) {
    const int kg = kb + (i < 8 ? i : i + 8); // 0..575
    const int j  = kg >> 6;                  // tap index
    const int c  = kg & 63;                  // channel slot within tap
    float w;
    if (c == 0) w = (j == 0) ? W[n * 568] : 0.0f;
    else        w = W[n * 568 + 1 + j * 63 + (c - 1)];
    __bf16 h, l;
    bf_split(w, h, l);
    const int o = (cb * 32 + lane) * 16 + i;
    Bhi[o] = h;
    Blo[o] = l;
  }
}

// ---------------------------------------------------------------------------
// Main kernel: 1024 blocks x 256 threads (8 wave32). Block = 128 pixels x 64 out.
// ---------------------------------------------------------------------------
__global__ __launch_bounds__(256)
void lorentz_conv2d_wmma(const float* __restrict__ xg,
                         const unsigned short* __restrict__ Bhi_u,
                         const unsigned short* __restrict__ Blo_u,
                         const float* __restrict__ bias,
                         float* __restrict__ out) {
  const __bf16* Bhi = (const __bf16*)Bhi_u;
  const __bf16* Blo = (const __bf16*)Blo_u;

  __shared__ __bf16 halo_hi[4 * 66 * HALO_STRIDE];
  __shared__ __bf16 halo_lo[4 * 66 * HALO_STRIDE];
  __shared__ __bf16 t_hi[128];
  __shared__ __bf16 t_lo[128];

  const int tid = threadIdx.x;
  const int bid = blockIdx.x;
  const int p0  = bid << 7;          // first pixel of tile
  const int b   = bid >> 5;          // batch (4096 px / image, 128 px / block)
  const int y0  = (bid << 1) & 63;   // first image row of tile (2 rows/block)

  // ---- stage fp32 halo (4 rows x 66 cols x 64 ch) as bf16 hi/lo in LDS ----
  for (int q = tid; q < 4 * 66; q += 256) {
    const int hr = q / 66;
    const int hc = q - hr * 66;
    const int yy = y0 - 1 + hr;
    const int xx = hc - 1;
    __bf16* dh = &halo_hi[(hr * 66 + hc) * HALO_STRIDE];
    __bf16* dl = &halo_lo[(hr * 66 + hc) * HALO_STRIDE];
    const bool inb = (yy >= 0) && (yy < 64) && (xx >= 0) && (xx < 64);
    const float* src = xg + (((long)b * 64 + yy) * 64 + xx) * 64;
#pragma unroll
    for (int c = 0; c < 64; c += 4) {
      v4f v = (v4f)0.0f;
      if (inb) v = *(const v4f*)(src + c);
#pragma unroll
      for (int e = 0; e < 4; ++e) bf_split(v[e], dh[c + e], dl[c + e]);
    }
  }

  // ---- t_rescaled = sqrt(sum_j max(x0_j,1)^2 - 8), fp32, one pixel/thread ----
  if (tid < 128) {
    const int y = y0 + (tid >> 6);
    const int x = tid & 63;
    float sum = 0.0f;
#pragma unroll
    for (int j = 0; j < 9; ++j) {
      const int yy = y + (j / 3) - 1;
      const int xx = x + (j % 3) - 1;
      float v = 0.0f;
      if (yy >= 0 && yy < 64 && xx >= 0 && xx < 64)
        v = xg[(((long)b * 64 + yy) * 64 + xx) * 64];
      v = fmaxf(v, 1.0f);
      sum = fmaf(v, v, sum);
    }
    bf_split(sqrtf(sum - 8.0f), t_hi[tid], t_lo[tid]);
  }
  __syncthreads();

  // ---- GEMM: each wave computes 16 rows x 64 cols via 18 K-chunks ----
  const int lane  = tid & 31;
  const int wave  = tid >> 5;
  const int n15   = lane & 15;
  const int hseli = lane >> 4;                // 0: K-runs {0-7,16-23}, 1: {8-15,24-31}
  const int mrow  = (wave << 4) + n15;        // A row this lane supplies (0..127)
  const int yoff  = mrow >> 6;                // row-in-tile (0/1)
  const int xm    = mrow & 63;

  v8f acc[4] = {};

#pragma unroll
  for (int chunk = 0; chunk < NCHUNK; ++chunk) {
    const int j  = chunk >> 1;                // tap 0..8
    const int h  = chunk & 1;                 // channel half
    const int hr = 1 + yoff + (j / 3) - 1;    // halo row 0..3
    const int hc = 1 + xm + (j % 3) - 1;      // halo col 0..65
    const int c0 = h * 32 + (hseli ? 8 : 0);

    const __bf16* pah = &halo_hi[(hr * 66 + hc) * HALO_STRIDE + c0];
    const __bf16* pal = &halo_lo[(hr * 66 + hc) * HALO_STRIDE + c0];
    v8bf ah0 = *(const v8bf*)pah;
    v8bf ah1 = *(const v8bf*)(pah + 16);
    v8bf al0 = *(const v8bf*)pal;
    v8bf al1 = *(const v8bf*)(pal + 16);
    v16bf ah = __builtin_shufflevector(ah0, ah1, 0,1,2,3,4,5,6,7,8,9,10,11,12,13,14,15);
    v16bf al = __builtin_shufflevector(al0, al1, 0,1,2,3,4,5,6,7,8,9,10,11,12,13,14,15);

    // tap-0 channel-0 slot carries t_rescaled (K=0 lives in element 0, lanes 0-15)
    if (chunk == 0 && hseli == 0) {
      ah[0] = t_hi[mrow];
      al[0] = t_lo[mrow];
    }

#pragma unroll
    for (int nb = 0; nb < 4; ++nb) {
      const long bofs = ((long)(chunk * 4 + nb) * 32 + lane) * 16;
      v16bf bh = *(const v16bf*)(Bhi + bofs);
      v16bf bl = *(const v16bf*)(Blo + bofs);
      acc[nb] = __builtin_amdgcn_wmma_f32_16x16x32_bf16(
          false, ah, false, bh, (short)0, acc[nb], false, false);
      acc[nb] = __builtin_amdgcn_wmma_f32_16x16x32_bf16(
          false, ah, false, bl, (short)0, acc[nb], false, false);
      acc[nb] = __builtin_amdgcn_wmma_f32_16x16x32_bf16(
          false, al, false, bh, (short)0, acc[nb], false, false);
    }
  }

  // ---- epilogue: bias, y_space stores, y_time = sqrt(sum y^2 + 1) ----
  float bv[4];
#pragma unroll
  for (int nb = 0; nb < 4; ++nb) bv[nb] = bias[nb * 16 + n15];

  const int rbase = hseli ? 8 : 0;            // C layout: lanes<16 own M=r, lanes>=16 own M=8+r
#pragma unroll
  for (int r = 0; r < 8; ++r) {
    const int m = (wave << 4) + rbase + r;
    float* orow = out + ((long)p0 + m) * 65;
    float ss = 0.0f;
#pragma unroll
    for (int nb = 0; nb < 4; ++nb) {
      const float y = acc[nb][r] + bv[nb];
      orow[1 + nb * 16 + n15] = y;
      ss = fmaf(y, y, ss);
    }
    // butterfly sum over the 16 lanes holding this row's 64 N-values
#pragma unroll
    for (int msk = 1; msk < 16; msk <<= 1)
      ss += __shfl_xor(ss, msk, 32);
    if (n15 == 0) orow[0] = sqrtf(ss + 1.0f);
  }
}

extern "C" void kernel_launch(void* const* d_in, const int* in_sizes, int n_in,
                              void* d_out, int out_size, void* d_ws, size_t ws_size,
                              hipStream_t stream) {
  const float* xg   = (const float*)d_in[0];   // [32,64,64,64] f32
  const float* W    = (const float*)d_in[1];   // [64,568] f32
  const float* bias = (const float*)d_in[2];   // [64] f32
  float* out = (float*)d_out;                  // [32,64,64,65] f32

  unsigned short* Bhi = (unsigned short*)d_ws;
  unsigned short* Blo = Bhi + B_ELEMS;         // 2 * 36864 bf16 = 144 KB total

  prep_B_kernel<<<NCHUNK * 4, 32, 0, stream>>>(W, Bhi, Blo);

  const int npix   = in_sizes[0] / 64;         // 131072
  const int nblk   = npix / 128;               // 1024
  lorentz_conv2d_wmma<<<nblk, 256, 0, stream>>>(xg, Bhi, Blo, bias, out);
}
